// GIDEncoder_17626545782864
// MI455X (gfx1250) — matrix operand order repaired and checked by reading, not compiled, not run on hardware
//
#include <hip/hip_runtime.h>
#include <hip/hip_bf16.h>

typedef __attribute__((ext_vector_type(16))) __bf16 v16bf;
typedef __attribute__((ext_vector_type(8)))  float  v8f;

union Frag { v16bf v; unsigned u[8]; uint4 q[2]; };

__device__ __forceinline__ unsigned short f2bf(float f) {
    union { float f; unsigned u; } c; c.f = f;
    unsigned u = c.u;
    unsigned r = u + 0x7fffu + ((u >> 16) & 1u);
    return (unsigned short)(r >> 16);
}
__device__ __forceinline__ unsigned pack2bf(float lo, float hi) {
    return (unsigned)f2bf(lo) | ((unsigned)f2bf(hi) << 16);
}

// A-fragment (16x32 bf16, MxK): lane m = lane&15, hi = lane>>4.
// VGPR p -> K = 2p + 16*hi => one contiguous 32B span at element offset hi*16.
__device__ __forceinline__ void load_fragA(Frag& f, const unsigned short* rowbase, int hi) {
    f.q[0] = *(const uint4*)(rowbase + hi * 16);
    f.q[1] = *(const uint4*)(rowbase + hi * 16 + 8);
}
// B-fragment (32x16 bf16, KxN): lane n = lane&15, hi = lane>>4.
// VGPR p -> K = 2p + 16*hi => one contiguous 32B span at element offset hi*16.
__device__ __forceinline__ void load_fragB(Frag& f, const unsigned short* colbase, int hi) {
    f.q[0] = *(const uint4*)(colbase + hi * 16);
    f.q[1] = *(const uint4*)(colbase + hi * 16 + 8);
}

#define BB   32
#define SS   512
#define NTOK (BB * SS)

// ---------------------------------------------------------------------------
// Stats over the 73 BN'd input features (obs32 | act8 | rew1 | trn32).
// ---------------------------------------------------------------------------
__global__ __launch_bounds__(256) void stats1_kernel(
    const float* __restrict__ obs, const float* __restrict__ act,
    const float* __restrict__ rew, float* __restrict__ stats)
{
    __shared__ float s1[256], s2[256];
    int f = blockIdx.x, tid = threadIdx.x;
    float sum = 0.f, sq = 0.f;
    for (int i = tid; i < NTOK; i += 256) {
        int b = i >> 9, s = i & 511;
        int prev = (b << 9) + (s == 0 ? 0 : s - 1);
        float v;
        if (f < 32)        v = obs[prev * 32 + f];
        else if (f < 40)   v = act[(size_t)i * 8 + (f - 32)];
        else if (f == 40)  v = rew[i];
        else { int ff = f - 41; v = obs[(size_t)i * 32 + ff] - obs[prev * 32 + ff]; }
        sum += v; sq += v * v;
    }
    s1[tid] = sum; s2[tid] = sq; __syncthreads();
    for (int o = 128; o >= 1; o >>= 1) {
        if (tid < o) { s1[tid] += s1[tid + o]; s2[tid] += s2[tid + o]; }
        __syncthreads();
    }
    if (tid == 0) {
        float m = s1[0] * (1.f / NTOK);
        float var = s2[0] * (1.f / NTOK) - m * m;
        stats[f] = m;
        stats[73 + f] = rsqrtf(var + 1e-5f);
    }
}

// ---------------------------------------------------------------------------
// Encoder: BN + 4 small projections -> enc[16384, 256] (f32)
// ---------------------------------------------------------------------------
__global__ __launch_bounds__(256) void encode_kernel(
    const float* __restrict__ obs, const float* __restrict__ act,
    const float* __restrict__ rew, const float* __restrict__ stats,
    const float* __restrict__ og, const float* __restrict__ ob,
    const float* __restrict__ ag, const float* __restrict__ ab,
    const float* __restrict__ rg, const float* __restrict__ rb,
    const float* __restrict__ tg, const float* __restrict__ tb,
    const float* __restrict__ W_obs, const float* __restrict__ b_obs,
    const float* __restrict__ W_act, const float* __restrict__ b_act,
    const float* __restrict__ W_rew, const float* __restrict__ b_rew,
    const float* __restrict__ W_trn, const float* __restrict__ b_trn,
    float* __restrict__ enc)
{
    __shared__ float xin[80];
    int tok = blockIdx.x;
    int b = tok >> 9, s = tok & 511;
    int prev = (b << 9) + (s == 0 ? 0 : s - 1);
    int tid = threadIdx.x;
    if (tid < 73) {
        float v, gg, bb2;
        if (tid < 32)      { v = obs[prev * 32 + tid];                 gg = og[tid];    bb2 = ob[tid]; }
        else if (tid < 40) { int f = tid - 32; v = act[(size_t)tok * 8 + f]; gg = ag[f]; bb2 = ab[f]; }
        else if (tid == 40){ v = rew[tok];                             gg = rg[0];      bb2 = rb[0]; }
        else               { int f = tid - 41;
                             v = obs[(size_t)tok * 32 + f] - obs[prev * 32 + f];
                             gg = tg[f]; bb2 = tb[f]; }
        xin[tid] = (v - stats[tid]) * stats[73 + tid] * gg + bb2;
    }
    __syncthreads();
    int seg = tid >> 6, j = tid & 63;
    float a = 0.f;
    if (seg == 0)      { for (int k = 0; k < 32; ++k) a += xin[k]      * W_obs[k * 64 + j]; a += b_obs[j]; }
    else if (seg == 1) { for (int k = 0; k < 8;  ++k) a += xin[32 + k] * W_act[k * 64 + j]; a += b_act[j]; }
    else if (seg == 2) { a = xin[40] * W_rew[j] + b_rew[j]; }
    else               { for (int k = 0; k < 32; ++k) a += xin[41 + k] * W_trn[k * 64 + j]; a += b_trn[j]; }
    enc[(size_t)tok * 256 + tid] = a;
}

// ---------------------------------------------------------------------------
// LayerNorm over 256 features; one wave per token.
// ---------------------------------------------------------------------------
__global__ __launch_bounds__(256) void ln_kernel(
    const float* __restrict__ enc, const float* __restrict__ g,
    const float* __restrict__ bta, float* __restrict__ out)
{
    int lane = threadIdx.x & 31, w = threadIdx.x >> 5;
    int tok = blockIdx.x * 8 + w;
    size_t base = (size_t)tok * 256;
    float v[8]; float sum = 0.f, sq = 0.f;
#pragma unroll
    for (int i = 0; i < 8; ++i) {
        v[i] = enc[base + i * 32 + lane];
        sum += v[i]; sq += v[i] * v[i];
    }
#pragma unroll
    for (int o = 16; o >= 1; o >>= 1) {
        sum += __shfl_xor(sum, o, 32);
        sq  += __shfl_xor(sq,  o, 32);
    }
    float mean = sum * (1.f / 256.f);
    float var  = sq * (1.f / 256.f) - mean * mean;
    float rstd = rsqrtf(var + 1e-5f);
#pragma unroll
    for (int i = 0; i < 8; ++i) {
        int col = i * 32 + lane;
        out[base + col] = (v[i] - mean) * rstd * g[col] + bta[col];
    }
}

// ---------------------------------------------------------------------------
// Pack [W_int | W_es] -> Wie[256,256]
// ---------------------------------------------------------------------------
__global__ __launch_bounds__(256) void pack_kernel(
    const float* __restrict__ wi, const float* __restrict__ we,
    float* __restrict__ wie)
{
    int i = blockIdx.x * 256 + threadIdx.x;
    if (i < 256 * 256) {
        int k = i >> 8, n = i & 255;
        wie[i] = (n < 128) ? wi[k * 128 + n] : we[k * 128 + (n - 128)];
    }
}

// ---------------------------------------------------------------------------
// Generic bf16 WMMA GEMM: C = act( (A*scaleK+shiftK) @ B + bias )
// A: MxK f32 row-major, B: KxN f32 row-major. Tile 128x64, K-step 32.
// Compile-time: SCALE (BN fold), ACT (0/1/2 = none/relu/tanh), OUTB (bf16 out).
// ---------------------------------------------------------------------------
template <bool SCALE, int ACT, bool OUTB>
__global__ __launch_bounds__(256) void gemm_wmma(
    const float* __restrict__ A, const float* __restrict__ Bw,
    const float* __restrict__ scaleK, const float* __restrict__ shiftK,
    const float* __restrict__ bias,
    float* __restrict__ outF, unsigned short* __restrict__ outB,
    int M, int N, int K)
{
    __shared__ unsigned short Alds[128 * 32];
    __shared__ unsigned short Blds[64 * 32];   // n-major: [n][k]
    int tid = threadIdx.x, lane = tid & 31, w = tid >> 5;
    int hi = lane >> 4, ln16 = lane & 15;
    int m0 = blockIdx.y * 128, n0 = blockIdx.x * 64;
    v8f acc[4] = {};

    for (int k0 = 0; k0 < K; k0 += 32) {
        // ---- stage A tile (128x32) : float4 loads, packed b64 LDS stores
#pragma unroll
        for (int t = 0; t < 4; ++t) {
            int idx4 = tid + t * 256;            // float4 index
            int r = idx4 >> 3, c = (idx4 & 7) * 4;
            float4 v = *(const float4*)&A[(size_t)(m0 + r) * K + k0 + c];
            if (SCALE) {
                float4 sc4 = *(const float4*)&scaleK[k0 + c];
                float4 sh4 = *(const float4*)&shiftK[k0 + c];
                v.x = v.x * sc4.x + sh4.x; v.y = v.y * sc4.y + sh4.y;
                v.z = v.z * sc4.z + sh4.z; v.w = v.w * sc4.w + sh4.w;
            }
            uint2 pk; pk.x = pack2bf(v.x, v.y); pk.y = pack2bf(v.z, v.w);
            *(uint2*)&Alds[r * 32 + c] = pk;
        }
        // ---- stage B tile (32k x 64n) transposed -> [n][k]
#pragma unroll
        for (int t = 0; t < 2; ++t) {
            int idx4 = tid + t * 256;            // float4 over n
            int k = idx4 >> 4, n = (idx4 & 15) * 4;
            float4 v = *(const float4*)&Bw[(size_t)(k0 + k) * N + n0 + n];
            Blds[(n + 0) * 32 + k] = f2bf(v.x);
            Blds[(n + 1) * 32 + k] = f2bf(v.y);
            Blds[(n + 2) * 32 + k] = f2bf(v.z);
            Blds[(n + 3) * 32 + k] = f2bf(v.w);
        }
        __syncthreads();

        Frag a;
        load_fragA(a, &Alds[((w << 4) + ln16) * 32], hi);
#pragma unroll
        for (int nt = 0; nt < 4; ++nt) {
            Frag bf;
            load_fragB(bf, &Blds[(nt * 16 + ln16) * 32], hi);
            acc[nt] = __builtin_amdgcn_wmma_f32_16x16x32_bf16(
                false, a.v, false, bf.v, (short)0, acc[nt], false, false);
        }
        __syncthreads();
    }

#pragma unroll
    for (int nt = 0; nt < 4; ++nt)
#pragma unroll
        for (int v = 0; v < 8; ++v) {
            int m = m0 + (w << 4) + v + 8 * hi;
            int n = n0 + nt * 16 + ln16;
            float val = acc[nt][v] + (bias ? bias[n] : 0.f);
            if (ACT == 1)      val = fmaxf(val, 0.f);
            else if (ACT == 2) val = tanhf(val);
            if (OUTB) outB[(size_t)m * N + n] = f2bf(val);
            else      outF[(size_t)m * N + n] = val;
        }
}

// ---------------------------------------------------------------------------
// Flash-style causal attention. qkv: bf16 [16384, 768] = [q|k|v], 4 heads x 64.
// Block = 4 waves; wave handles one 16-row q-tile of one (b,h).
// ---------------------------------------------------------------------------
__global__ __launch_bounds__(128) void attn_kernel(
    const unsigned short* __restrict__ qkv, float* __restrict__ x)
{
    int lane = threadIdx.x & 31, w = threadIdx.x >> 5;
    int hi = lane >> 4, ln16 = lane & 15;
    int bh = blockIdx.y, b = bh >> 2, h = bh & 3;
    int q0 = blockIdx.x * 64 + w * 16;
    __shared__ unsigned short Plds[4][16 * 32];
    unsigned short* pl = Plds[w];

    Frag qf[2];
    {
        const unsigned short* qb =
            qkv + ((size_t)(b * SS) + q0 + ln16) * 768 + h * 64;
#pragma unroll
        for (int dc = 0; dc < 2; ++dc) load_fragA(qf[dc], qb + dc * 32, hi);
    }
    v8f acc[4] = {};
    float rowm[8], rows[8];
#pragma unroll
    for (int v = 0; v < 8; ++v) { rowm[v] = -__builtin_inff(); rows[v] = 0.f; }
    const float sc = 0.125f;   // 1/sqrt(64)
    int nch = (q0 + 15) / 32 + 1;

    for (int c = 0; c < nch; ++c) {
        int kt0 = c * 32;
        v8f s[2];
#pragma unroll
        for (int sub = 0; sub < 2; ++sub) {
            v8f z = {}; s[sub] = z;
            const unsigned short* kb =
                qkv + ((size_t)(b * SS) + kt0 + sub * 16 + ln16) * 768 + 256 + h * 64;
#pragma unroll
            for (int dc = 0; dc < 2; ++dc) {
                Frag kf;
                load_fragB(kf, kb + dc * 32, hi);
                s[sub] = __builtin_amdgcn_wmma_f32_16x16x32_bf16(
                    false, qf[dc].v, false, kf.v, (short)0, s[sub], false, false);
            }
#pragma unroll
            for (int v = 0; v < 8; ++v) {
                int m = q0 + v + 8 * hi;
                int kk = kt0 + sub * 16 + ln16;
                float sv = s[sub][v] * sc;
                s[sub][v] = (kk > m) ? -__builtin_inff() : sv;
            }
        }
        float scl[8];
#pragma unroll
        for (int v = 0; v < 8; ++v) {
            float mv = fmaxf(s[0][v], s[1][v]);
#pragma unroll
            for (int o = 8; o >= 1; o >>= 1) mv = fmaxf(mv, __shfl_xor(mv, o, 16));
            float nm = fmaxf(rowm[v], mv);
            bool dead = (nm == -__builtin_inff());
            float sl = dead ? 1.f : __expf(rowm[v] - nm);
            float p0 = dead ? 0.f : __expf(s[0][v] - nm);
            float p1 = dead ? 0.f : __expf(s[1][v] - nm);
            float ps = p0 + p1;
#pragma unroll
            for (int o = 8; o >= 1; o >>= 1) ps += __shfl_xor(ps, o, 16);
            rows[v] = rows[v] * sl + ps;
            rowm[v] = nm;
            scl[v] = sl;
            pl[(v + 8 * hi) * 32 + ln16]      = f2bf(p0);
            pl[(v + 8 * hi) * 32 + 16 + ln16] = f2bf(p1);
        }
#pragma unroll
        for (int nt = 0; nt < 4; ++nt)
#pragma unroll
            for (int v = 0; v < 8; ++v) acc[nt][v] *= scl[v];

        asm volatile("s_wait_dscnt 0" ::: "memory");   // wave-local LDS RAW

        Frag pf;
        load_fragA(pf, &pl[ln16 * 32], hi);
#pragma unroll
        for (int nt = 0; nt < 4; ++nt) {
            Frag vf;
            int d = nt * 16 + ln16;
#pragma unroll
            for (int p = 0; p < 8; ++p) {
                int k1 = kt0 + p * 2 + hi * 16;
                size_t i0 = ((size_t)(b * SS) + k1) * 768 + 512 + h * 64 + d;
                unsigned e0 = qkv[i0];
                unsigned e1 = qkv[i0 + 768];
                vf.u[p] = e0 | (e1 << 16);
            }
            acc[nt] = __builtin_amdgcn_wmma_f32_16x16x32_bf16(
                false, pf.v, false, vf.v, (short)0, acc[nt], false, false);
        }
    }
#pragma unroll
    for (int nt = 0; nt < 4; ++nt)
#pragma unroll
        for (int v = 0; v < 8; ++v) {
            int m = q0 + v + 8 * hi;
            int n = nt * 16 + ln16;
            x[((size_t)(b * SS) + m) * 512 + h * 64 + n] = acc[nt][v] / rows[v];
        }
}

// ---------------------------------------------------------------------------
// Integrator (cumsum) + exponential-smoothing heads -> x[:, 256:512]
// ---------------------------------------------------------------------------
__global__ __launch_bounds__(256) void intes_kernel(
    const float* __restrict__ ie, float* __restrict__ x)
{
    int t = blockIdx.x * 256 + threadIdx.x;
    int b = t >> 8, col = t & 255;
    size_t base = (size_t)b * SS;
    if (col < 128) {
        float run = 0.f;
        for (int s = 0; s < SS; ++s) {
            run += ie[(base + s) * 256 + col];
            x[(base + s) * 512 + 256 + col] = run;
        }
    } else {
        float alpha = (col < 192) ? 0.1f : 0.25f;
        float omc = 1.f - alpha;
        float n = 0.f, pw = 1.f;
        for (int s = 0; s < SS; ++s) {
            n = alpha * ie[(base + s) * 256 + col] + omc * n;
            pw *= omc;
            x[(base + s) * 512 + 256 + col] = n / (1.f - pw);
        }
    }
}

// ---------------------------------------------------------------------------
// BN stats over x (512 features), then fold to scale/shift
// ---------------------------------------------------------------------------
__global__ __launch_bounds__(256) void stats2_kernel(
    const float* __restrict__ x, float* __restrict__ stats)
{
    __shared__ float s1[256], s2[256];
    int f = blockIdx.x, tid = threadIdx.x;
    float sum = 0.f, sq = 0.f;
    for (int i = tid; i < NTOK; i += 256) {
        float v = x[(size_t)i * 512 + f];
        sum += v; sq += v * v;
    }
    s1[tid] = sum; s2[tid] = sq; __syncthreads();
    for (int o = 128; o >= 1; o >>= 1) {
        if (tid < o) { s1[tid] += s1[tid + o]; s2[tid] += s2[tid + o]; }
        __syncthreads();
    }
    if (tid == 0) {
        float m = s1[0] * (1.f / NTOK);
        float var = s2[0] * (1.f / NTOK) - m * m;
        stats[f] = m;
        stats[512 + f] = rsqrtf(var + 1e-5f);
    }
}

__global__ __launch_bounds__(256) void fold_kernel(
    const float* __restrict__ stats, const float* __restrict__ g,
    const float* __restrict__ bta, float* __restrict__ scale,
    float* __restrict__ shift)
{
    int f = blockIdx.x * 256 + threadIdx.x;
    if (f < 512) {
        float m = stats[f], r = stats[512 + f];
        float sc = g[f] * r;
        scale[f] = sc;
        shift[f] = bta[f] - m * sc;
    }
}

// ---------------------------------------------------------------------------
extern "C" void kernel_launch(void* const* d_in, const int* in_sizes, int n_in,
                              void* d_out, int out_size, void* d_ws, size_t ws_size,
                              hipStream_t stream)
{
    const float* obs_seq = (const float*)d_in[0];
    const float* act_seq = (const float*)d_in[1];
    const float* rew_seq = (const float*)d_in[2];
    const float* W_obs   = (const float*)d_in[3];
    const float* b_obs   = (const float*)d_in[4];
    const float* W_act   = (const float*)d_in[5];
    const float* b_act   = (const float*)d_in[6];
    const float* W_rew   = (const float*)d_in[7];
    const float* b_rew   = (const float*)d_in[8];
    const float* W_trn   = (const float*)d_in[9];
    const float* b_trn   = (const float*)d_in[10];
    const float* obs_g   = (const float*)d_in[11];
    const float* obs_b   = (const float*)d_in[12];
    const float* act_g   = (const float*)d_in[13];
    const float* act_b   = (const float*)d_in[14];
    const float* rew_g   = (const float*)d_in[15];
    const float* rew_b   = (const float*)d_in[16];
    const float* trn_g   = (const float*)d_in[17];
    const float* trn_b   = (const float*)d_in[18];
    const float* ln_g    = (const float*)d_in[19];
    const float* ln_b    = (const float*)d_in[20];
    const float* W_attn  = (const float*)d_in[21];
    const float* W_int   = (const float*)d_in[22];
    const float* W_es    = (const float*)d_in[23];
    const float* bn_g    = (const float*)d_in[24];
    const float* bn_b    = (const float*)d_in[25];
    const float* W_dec1  = (const float*)d_in[26];
    const float* b_dec1  = (const float*)d_in[27];
    const float* W_dec2  = (const float*)d_in[28];
    const float* b_dec2  = (const float*)d_in[29];
    float* out = (float*)d_out;

    // workspace carve; aliasing reuses dead regions
    float* stats1 = (float*)d_ws;               // 256
    float* stats2 = stats1 + 256;               // 1024
    float* scale2 = stats2 + 1024;              // 512
    float* shift2 = scale2 + 512;               // 512
    float* Wie    = shift2 + 512;               // 65536
    float* enc    = Wie + 65536;                // 16384*256
    float* lnb    = enc + (size_t)NTOK * 256;   // 16384*256
    float* xb     = enc;                        // alias enc+ln (dead after GEMMs)
    unsigned short* qkv = (unsigned short*)(lnb + (size_t)NTOK * 256); // 16384*768 bf16
    float* h1     = (float*)qkv;                // alias qkv (dead after attention)
    float* ie     = (float*)(qkv + (size_t)NTOK * 768);               // 16384*256 f32

    stats1_kernel<<<73, 256, 0, stream>>>(obs_seq, act_seq, rew_seq, stats1);

    encode_kernel<<<NTOK, 256, 0, stream>>>(
        obs_seq, act_seq, rew_seq, stats1,
        obs_g, obs_b, act_g, act_b, rew_g, rew_b, trn_g, trn_b,
        W_obs, b_obs, W_act, b_act, W_rew, b_rew, W_trn, b_trn, enc);

    ln_kernel<<<NTOK / 8, 256, 0, stream>>>(enc, ln_g, ln_b, lnb);
    pack_kernel<<<256, 256, 0, stream>>>(W_int, W_es, Wie);

    // qkv = ln(enc) @ W_attn   (bf16 out)
    gemm_wmma<false, 0, true><<<dim3(768 / 64, NTOK / 128), 256, 0, stream>>>(
        lnb, W_attn, nullptr, nullptr, nullptr, nullptr, qkv, NTOK, 768, 256);
    // ie = enc @ [W_int|W_es]  (f32 out)
    gemm_wmma<false, 0, false><<<dim3(256 / 64, NTOK / 128), 256, 0, stream>>>(
        enc, Wie, nullptr, nullptr, nullptr, ie, nullptr, NTOK, 256, 256);

    attn_kernel<<<dim3(SS / 64, BB * 4), 128, 0, stream>>>(qkv, xb);
    intes_kernel<<<32, 256, 0, stream>>>(ie, xb);

    stats2_kernel<<<512, 256, 0, stream>>>(xb, stats2);
    fold_kernel<<<2, 256, 0, stream>>>(stats2, bn_g, bn_b, scale2, shift2);

    // h1 = relu( bn(x) @ W_dec1 + b_dec1 )
    gemm_wmma<true, 1, false><<<dim3(256 / 64, NTOK / 128), 256, 0, stream>>>(
        xb, W_dec1, scale2, shift2, b_dec1, h1, nullptr, NTOK, 256, 512);
    // out = tanh( h1 @ W_dec2 + b_dec2 )
    gemm_wmma<false, 2, false><<<dim3(128 / 64, NTOK / 128), 256, 0, stream>>>(
        h1, W_dec2, nullptr, nullptr, b_dec2, out, nullptr, NTOK, 128, 256);
}
